// AttentionBlock_86062554677928
// MI455X (gfx1250) — matrix operand order repaired and checked by reading, not compiled, not run on hardware
//
#include <hip/hip_runtime.h>

// ---------------------------------------------------------------------------
// AttentionBlock for MI455X (gfx1250): GroupNorm -> QKV GEMM -> flash attn
// -> proj GEMM + residual.  All matmuls via v_wmma_f32_16x16x32_f16.
// Register-blocked GEMMs (64x32 per wave), LDS-staged K/V in attention.
// ---------------------------------------------------------------------------

typedef __attribute__((ext_vector_type(16))) _Float16 v16h;
typedef __attribute__((ext_vector_type(8)))  _Float16 v8h;
typedef __attribute__((ext_vector_type(8)))  float    v8f;

#define NB 8        // batch
#define NC 512      // channels
#define NT 1024     // tokens (32*32)
#define NH 8        // heads
#define CH 64       // channels per head
#define SCALE 0.35355339059327373f   // 64^-0.25

union V16U { v16h v; v8h h[2]; };

__device__ inline v8f zero8() {
  v8f z = {0.f, 0.f, 0.f, 0.f, 0.f, 0.f, 0.f, 0.f};
  return z;
}

// A-fragment (16x32, M x K) for row `rowptr` (row-major, contiguous K).
// Element e of lane l holds K = kb + 16*(l/16) + e.
__device__ inline v16h load_afrag(const _Float16* __restrict__ rowptr, int kb, int g) {
  V16U r;
  r.h[0] = *(const v8h*)(rowptr + kb + 16 * g);
  r.h[1] = *(const v8h*)(rowptr + kb + 16 * g + 8);
  return r.v;
}

// B-fragment (32x16, K x N): lane = N column; element e holds K = kb + 16*(l/16) + e.
// `colptr` points at the start of this lane's N-column storage (contiguous in K).
__device__ inline v16h load_bfrag(const _Float16* __restrict__ colptr, int kb, int g) {
  return *(const v16h*)(colptr + kb + 16 * g);
}

__device__ inline v8f wmma_f16(v16h a, v16h b, v8f c) {
  return __builtin_amdgcn_wmma_f32_16x16x32_f16(
      /*neg_a=*/false, a, /*neg_b=*/false, b,
      /*c_mod=*/(short)0, c, /*reuse_a=*/false, /*reuse_b=*/false);
}

// ---------------------------------------------------------------------------
// fp32 -> f16 weight conversion
// ---------------------------------------------------------------------------
__global__ __launch_bounds__(256) void cvt_kernel(const float* __restrict__ a,
                                                  _Float16* __restrict__ o, int n) {
  int i = blockIdx.x * 256 + threadIdx.x;
  if (i < n) o[i] = (_Float16)a[i];
}

// ---------------------------------------------------------------------------
// GroupNorm over (C/32 = 16 channels) x T, write t-major f16 xnT[b][t][c]
// ---------------------------------------------------------------------------
__global__ __launch_bounds__(256) void gnorm_kernel(const float* __restrict__ x,
                                                    const float* __restrict__ w,
                                                    const float* __restrict__ bvec,
                                                    _Float16* __restrict__ xnT) {
  __shared__ float s1[256], s2[256];
  int b = blockIdx.x >> 5;    // 32 groups per batch
  int grp = blockIdx.x & 31;
  int c0 = grp * 16;
  const float* xb = x + ((size_t)b * NC + c0) * NT;

  float sum = 0.f, sq = 0.f;
  for (int idx = threadIdx.x; idx < 16 * NT; idx += 256) {
    float v = xb[idx];
    sum += v; sq += v * v;
  }
  s1[threadIdx.x] = sum; s2[threadIdx.x] = sq;
  __syncthreads();
  for (int off = 128; off > 0; off >>= 1) {
    if (threadIdx.x < off) {
      s1[threadIdx.x] += s1[threadIdx.x + off];
      s2[threadIdx.x] += s2[threadIdx.x + off];
    }
    __syncthreads();
  }
  float mean = s1[0] * (1.f / 16384.f);
  float var  = s2[0] * (1.f / 16384.f) - mean * mean;
  float rstd = rsqrtf(var + 1e-5f);

  for (int idx = threadIdx.x; idx < 16 * NT; idx += 256) {
    int ci = idx >> 10, t = idx & 1023;
    int c = c0 + ci;
    float y = (xb[idx] - mean) * rstd * w[c] + bvec[c];
    xnT[((size_t)b * NT + t) * NC + c] = (_Float16)y;
  }
}

// ---------------------------------------------------------------------------
// QKV GEMM: qkv[b,o,t] = sum_c W[o,c]*xn[b,c,t] + bias[o]; route to q/k/v.
// Register-blocked: one wave computes 64(M=o) x 32(N=t) = 4x2 WMMA tiles.
// q,k stored t-major [bh][t][c] pre-scaled; v stored c-major [bh][c][s].
// 8b * 24mt * 32nt = 6144 waves = 768 blocks.
// ---------------------------------------------------------------------------
__global__ __launch_bounds__(256) void qkv_gemm_kernel(
    const _Float16* __restrict__ W, const _Float16* __restrict__ xnT,
    const float* __restrict__ bias,
    _Float16* __restrict__ qT, _Float16* __restrict__ kT,
    _Float16* __restrict__ vv) {
  int lane = threadIdx.x & 31, g = lane >> 4, ln = lane & 15;
  int wid = blockIdx.x * 8 + (threadIdx.x >> 5);
  int b = wid / (24 * 32);
  int rem = wid % (24 * 32);
  int mt = rem / 32, nt = rem % 32;

  const _Float16* wrow[4];
#pragma unroll
  for (int mi = 0; mi < 4; mi++)
    wrow[mi] = W + (size_t)(mt * 64 + mi * 16 + ln) * NC;
  const _Float16* xb[2];
#pragma unroll
  for (int ns = 0; ns < 2; ns++)
    xb[ns] = xnT + ((size_t)b * NT + nt * 32 + ns * 16 + ln) * NC;

  v8f acc[4][2];
#pragma unroll
  for (int mi = 0; mi < 4; mi++)
#pragma unroll
    for (int ns = 0; ns < 2; ns++) acc[mi][ns] = zero8();

#pragma unroll 4
  for (int kb = 0; kb < NC; kb += 32) {
    v16h af[4], bf[2];
#pragma unroll
    for (int mi = 0; mi < 4; mi++) af[mi] = load_afrag(wrow[mi], kb, g);
#pragma unroll
    for (int ns = 0; ns < 2; ns++) bf[ns] = load_bfrag(xb[ns], kb, g);
#pragma unroll
    for (int mi = 0; mi < 4; mi++)
#pragma unroll
      for (int ns = 0; ns < 2; ns++)
        acc[mi][ns] = wmma_f16(af[mi], bf[ns], acc[mi][ns]);
  }

#pragma unroll
  for (int mi = 0; mi < 4; mi++) {
    int ot = mt * 64 + mi * 16;
    int head = ot / 192;
    int r = ot - head * 192;
    int sel = r >> 6;          // 0=q 1=k 2=v (uniform per 16-row subtile)
    int cb = r & 63;
    int bh = b * NH + head;
#pragma unroll
    for (int ns = 0; ns < 2; ns++) {
      int t = nt * 32 + ns * 16 + ln;
#pragma unroll
      for (int i = 0; i < 8; i++) {
        int o = ot + i + 8 * g;
        int c = cb + i + 8 * g;
        float val = acc[mi][ns][i] + bias[o];
        if (sel == 0)
          qT[((size_t)bh * NT + t) * CH + c] = (_Float16)(val * SCALE);
        else if (sel == 1)
          kT[((size_t)bh * NT + t) * CH + c] = (_Float16)(val * SCALE);
        else
          vv[((size_t)bh * CH + c) * NT + t] = (_Float16)val;
      }
    }
  }
}

// ---------------------------------------------------------------------------
// Flash attention: block = one (b,h) x 128 t-rows (8 waves, one 16-row tile
// each).  K/V tiles for each 32-wide s-block are cooperatively staged in LDS
// and shared by all 8 waves.  Online softmax; output aT[b][t][h*64+cv] (f16).
// 64 bh * 8 groups = 512 blocks.
// ---------------------------------------------------------------------------
__global__ __launch_bounds__(256) void attn_kernel(
    const _Float16* __restrict__ qT, const _Float16* __restrict__ kT,
    const _Float16* __restrict__ vv, _Float16* __restrict__ aT) {
  __shared__ __align__(16) _Float16 lK[32 * 64];      // [s_local][c]
  __shared__ __align__(16) _Float16 lV[64 * 32];      // [cv][s_local]
  __shared__ __align__(16) _Float16 lP[8][16 * 32];   // per-wave P transpose
  int wib = threadIdx.x >> 5;
  int lane = threadIdx.x & 31, g = lane >> 4, ln = lane & 15;
  int bh = blockIdx.x >> 3;
  int ttile = (blockIdx.x & 7) * 8 + wib;
  int b = bh >> 3, h = bh & 7;

  const _Float16* qbase = qT + (size_t)bh * NT * CH;
  const _Float16* kbase = kT + (size_t)bh * NT * CH;
  const _Float16* vbase = vv + (size_t)bh * CH * NT;
  _Float16* lw = lP[wib];

  int trow = ttile * 16 + ln;
  v16h aq0 = load_afrag(qbase + (size_t)trow * CH, 0, g);
  v16h aq1 = load_afrag(qbase + (size_t)trow * CH, 32, g);

  // cooperative K/V staging indices (256 threads, 8 halves each)
  int kr = threadIdx.x >> 3, kc8 = (threadIdx.x & 7) * 8;   // K: 32 rows x 64
  int vr = threadIdx.x >> 2, vs8 = (threadIdx.x & 3) * 8;   // V: 64 rows x 32

  float m[8], ssum[8];
  v8f O[4];
#pragma unroll
  for (int i = 0; i < 8; i++) { m[i] = -1e30f; ssum[i] = 0.f; }
#pragma unroll
  for (int j = 0; j < 4; j++) O[j] = zero8();

  for (int sb = 0; sb < NT; sb += 32) {
    *(v8h*)(lK + kr * 64 + kc8) =
        *(const v8h*)(kbase + (size_t)(sb + kr) * CH + kc8);
    *(v8h*)(lV + vr * 32 + vs8) =
        *(const v8h*)(vbase + (size_t)vr * NT + sb + vs8);
    __syncthreads();

    const _Float16* k0 = lK + (size_t)ln * 64;
    const _Float16* k1 = lK + (size_t)(16 + ln) * 64;
    v8f S0 = zero8(), S1 = zero8();
    S0 = wmma_f16(aq0, *(const v16h*)(k0 + 16 * g), S0);
    S0 = wmma_f16(aq1, *(const v16h*)(k0 + 32 + 16 * g), S0);
    S1 = wmma_f16(aq0, *(const v16h*)(k1 + 16 * g), S1);
    S1 = wmma_f16(aq1, *(const v16h*)(k1 + 32 + 16 * g), S1);

    float P0[8], P1[8], cf[8];
#pragma unroll
    for (int i = 0; i < 8; i++) {
      float tm = fmaxf(S0[i], S1[i]);
      tm = fmaxf(tm, __shfl_xor(tm, 1));
      tm = fmaxf(tm, __shfl_xor(tm, 2));
      tm = fmaxf(tm, __shfl_xor(tm, 4));
      tm = fmaxf(tm, __shfl_xor(tm, 8));
      float nm = fmaxf(m[i], tm);
      cf[i] = __expf(m[i] - nm);
      m[i] = nm;
      float p0 = __expf(S0[i] - nm);
      float p1 = __expf(S1[i] - nm);
      P0[i] = p0; P1[i] = p1;
      float rs = p0 + p1;
      rs += __shfl_xor(rs, 1);
      rs += __shfl_xor(rs, 2);
      rs += __shfl_xor(rs, 4);
      rs += __shfl_xor(rs, 8);
      ssum[i] = ssum[i] * cf[i] + rs;
    }
#pragma unroll
    for (int j = 0; j < 4; j++)
#pragma unroll
      for (int i = 0; i < 8; i++) O[j][i] *= cf[i];

    // Transpose P: C/D layout -> A-fragment layout via per-wave LDS bounce.
    // DS ops from one wave execute in order, so no barrier needed here.
#pragma unroll
    for (int i = 0; i < 8; i++) {
      lw[(i + 8 * g) * 32 + ln]      = (_Float16)P0[i];
      lw[(i + 8 * g) * 32 + 16 + ln] = (_Float16)P1[i];
    }
    __asm__ volatile("" ::: "memory");
    V16U ap;
    ap.h[0] = *(const v8h*)(lw + ln * 32 + 16 * g);
    ap.h[1] = *(const v8h*)(lw + ln * 32 + 16 * g + 8);
    __asm__ volatile("" ::: "memory");

#pragma unroll
    for (int j = 0; j < 4; j++) {
      v16h bv = *(const v16h*)(lV + (size_t)(j * 16 + ln) * 32 + 16 * g);
      O[j] = wmma_f16(ap.v, bv, O[j]);
    }
    __syncthreads();   // protect lK/lV before next iteration's overwrite
  }

  _Float16* aout = aT + (size_t)b * NT * NC + (size_t)h * CH;
#pragma unroll
  for (int i = 0; i < 8; i++) {
    float inv = 1.f / ssum[i];
    int t = ttile * 16 + i + 8 * g;
#pragma unroll
    for (int j = 0; j < 4; j++)
      aout[(size_t)t * NC + j * 16 + ln] = (_Float16)(O[j][i] * inv);
  }
}

// ---------------------------------------------------------------------------
// Projection GEMM + bias + residual:
//   out[b,o,t] = sum_c W[o,c]*a[b,c,t] + pb[o] + x[b,o,t]
// Register-blocked 64x32 per wave.  8b * 8mt * 32nt = 2048 waves = 256 blocks.
// ---------------------------------------------------------------------------
__global__ __launch_bounds__(256) void proj_gemm_kernel(
    const _Float16* __restrict__ W, const _Float16* __restrict__ aT,
    const float* __restrict__ bias, const float* __restrict__ x,
    float* __restrict__ out) {
  int lane = threadIdx.x & 31, g = lane >> 4, ln = lane & 15;
  int wid = blockIdx.x * 8 + (threadIdx.x >> 5);
  int b = wid / (8 * 32);
  int rem = wid % (8 * 32);
  int mt = rem / 32, nt = rem % 32;

  const _Float16* wrow[4];
#pragma unroll
  for (int mi = 0; mi < 4; mi++)
    wrow[mi] = W + (size_t)(mt * 64 + mi * 16 + ln) * NC;
  const _Float16* ab[2];
#pragma unroll
  for (int ns = 0; ns < 2; ns++)
    ab[ns] = aT + ((size_t)b * NT + nt * 32 + ns * 16 + ln) * NC;

  v8f acc[4][2];
#pragma unroll
  for (int mi = 0; mi < 4; mi++)
#pragma unroll
    for (int ns = 0; ns < 2; ns++) acc[mi][ns] = zero8();

#pragma unroll 4
  for (int kb = 0; kb < NC; kb += 32) {
    v16h af[4], bf[2];
#pragma unroll
    for (int mi = 0; mi < 4; mi++) af[mi] = load_afrag(wrow[mi], kb, g);
#pragma unroll
    for (int ns = 0; ns < 2; ns++) bf[ns] = load_bfrag(ab[ns], kb, g);
#pragma unroll
    for (int mi = 0; mi < 4; mi++)
#pragma unroll
      for (int ns = 0; ns < 2; ns++)
        acc[mi][ns] = wmma_f16(af[mi], bf[ns], acc[mi][ns]);
  }

#pragma unroll
  for (int mi = 0; mi < 4; mi++) {
#pragma unroll
    for (int ns = 0; ns < 2; ns++) {
      int t = nt * 32 + ns * 16 + ln;
#pragma unroll
      for (int i = 0; i < 8; i++) {
        int o = mt * 64 + mi * 16 + i + 8 * g;
        size_t idx = ((size_t)b * NC + o) * NT + t;
        out[idx] = acc[mi][ns][i] + bias[o] + x[idx];
      }
    }
  }
}

// ---------------------------------------------------------------------------
extern "C" void kernel_launch(void* const* d_in, const int* in_sizes, int n_in,
                              void* d_out, int out_size, void* d_ws, size_t ws_size,
                              hipStream_t stream) {
  (void)in_sizes; (void)n_in; (void)out_size; (void)ws_size;
  const float* x      = (const float*)d_in[0];
  const float* norm_w = (const float*)d_in[1];
  const float* norm_b = (const float*)d_in[2];
  const float* qkv_w  = (const float*)d_in[3];
  const float* qkv_b  = (const float*)d_in[4];
  const float* proj_w = (const float*)d_in[5];
  const float* proj_b = (const float*)d_in[6];
  float* out = (float*)d_out;

  // Workspace layout (f16 elements). aT aliases xnT (xnT dead after qkv_gemm).
  _Float16* ws   = (_Float16*)d_ws;
  _Float16* xnT  = ws;                           // 8*1024*512 = 4,194,304
  _Float16* qW16 = xnT + (size_t)NB * NT * NC;   // 1536*512  =   786,432
  _Float16* pW16 = qW16 + 1536 * NC;             // 512*512   =   262,144
  _Float16* qT   = pW16 + NC * NC;               // 4,194,304
  _Float16* kT   = qT + (size_t)NB * NT * NC;    // 4,194,304
  _Float16* vv   = kT + (size_t)NB * NT * NC;    // 4,194,304
  _Float16* aT   = xnT;                          // alias (safe: serialized)

  cvt_kernel<<<(1536 * NC + 255) / 256, 256, 0, stream>>>(qkv_w, qW16, 1536 * NC);
  cvt_kernel<<<(NC * NC + 255) / 256, 256, 0, stream>>>(proj_w, pW16, NC * NC);
  gnorm_kernel<<<NB * 32, 256, 0, stream>>>(x, norm_w, norm_b, xnT);
  qkv_gemm_kernel<<<768, 256, 0, stream>>>(qW16, xnT, qkv_b, qT, kT, vv);
  attn_kernel<<<512, 256, 0, stream>>>(qT, kT, vv, aT);
  proj_gemm_kernel<<<256, 256, 0, stream>>>(pW16, aT, proj_b, x, out);
}
